// WindowAttention_63050119905817
// MI455X (gfx1250) — compile-verified, matrix operand order checked by reading, not run on hardware
//
#include <hip/hip_runtime.h>
#include <hip/hip_bf16.h>
#include <cstddef>

typedef __attribute__((ext_vector_type(16))) _Float16 v16h;
typedef __attribute__((ext_vector_type(8)))  float    v8f;

#define L_TOK 49
#define DIM   128
#define HEADS 4
#define NW    64

// ---------------- LDS layout (bytes), 148 KB arena with phase reuse --------
#define OFF_XA     0                  // 16 KB : x A-fragments (phase 0-1)
#define OFF_AATTN  0                  // 32 KB : attn A-fragments (phase 5+)
#define OFF_QK     32768              // 66 KB : q,k f32 [64][264] (phase 1-2)
#define QK_STR     264
#define OFF_S      32768              // 64 KB : S f32 [4][64][64] (phase 3-5)
#define OFF_OUTF   32768              // 33 KB : P f32 [64][132] (phase 6-7)
#define OUTF_STR   132
#define OFF_VB     100352             // 16 KB : v B-frags (written in phase 1!)
#define OFF_SCALES 116736             // 2 KB  : q/k inv-norms [2][4][64]
#define OFF_QA     118784             // 16 KB : q-hat A-frags, reused as proj A-frags
#define OFF_KB     135168             // 16 KB : k-hat B-frags
#define SMEM_BYTES 151552

// ---------------- workspace layout (bytes) ---------------------------------
#define WS_WQKV   0                   // 96 frags * 512 f16
#define WS_WPROJ  98304               // 32 frags * 512 f16
#define WS_B169   131072              // 169*4 f32
#define WS_CMB    135168              // 64*4*64*64 f32 = 4 MB padded bias+mask (transposed)

static __device__ __forceinline__ v8f wmma16(v16h a, v16h b, v8f c) {
  return __builtin_amdgcn_wmma_f32_16x16x32_f16(false, a, false, b, (short)0, c,
                                                false, false);
}

// ============ prep 1: repack qkv_w / proj_w into f16 B-fragments ============
__global__ __launch_bounds__(256) void pack_weights(
    const float* __restrict__ qkv_w, const float* __restrict__ proj_w,
    _Float16* __restrict__ wq, _Float16* __restrict__ wp) {
  int s = blockIdx.x * 256 + threadIdx.x;      // 128 frags * 32 slots = 4096
  int frag = s >> 5, ln = s & 31;
  int hi = ln >> 4, col = ln & 15;
  const float* W;
  _Float16* dst;
  int n, kc;
  if (frag < 96) { n = frag >> 2; kc = frag & 3; W = qkv_w;  dst = wq + (size_t)frag * 512 + ln * 16; }
  else { int f = frag - 96; n = f >> 2; kc = f & 3; W = proj_w; dst = wp + (size_t)f * 512 + ln * 16; }
  int N = n * 16 + col;
  const float* src = W + (size_t)N * 128 + kc * 32 + hi * 8;
  float4 a0 = *(const float4*)(src);
  float4 a1 = *(const float4*)(src + 4);
  float4 b0 = *(const float4*)(src + 16);
  float4 b1 = *(const float4*)(src + 20);
  v16h t;
  t[0]=(_Float16)a0.x; t[1]=(_Float16)a0.y; t[2]=(_Float16)a0.z; t[3]=(_Float16)a0.w;
  t[4]=(_Float16)a1.x; t[5]=(_Float16)a1.y; t[6]=(_Float16)a1.z; t[7]=(_Float16)a1.w;
  t[8]=(_Float16)b0.x; t[9]=(_Float16)b0.y; t[10]=(_Float16)b0.z; t[11]=(_Float16)b0.w;
  t[12]=(_Float16)b1.x; t[13]=(_Float16)b1.y; t[14]=(_Float16)b1.z; t[15]=(_Float16)b1.w;
  *(v16h*)dst = t;
}

// ============ prep 2: rpb MLP 169x2 -> 512 relu -> 4 ========================
__global__ __launch_bounds__(256) void rpb_mlp(
    const float* __restrict__ rc, const float* __restrict__ w1,
    const float* __restrict__ b1, const float* __restrict__ w2,
    const float* __restrict__ b2, float* __restrict__ b169) {
  int t = threadIdx.x;
  if (t >= 169) return;
  float c0 = rc[t * 2 + 0], c1 = rc[t * 2 + 1];
  float acc[HEADS] = {0.f, 0.f, 0.f, 0.f};
  for (int j = 0; j < 512; ++j) {
    float h = fmaxf(c0 * w1[j * 2 + 0] + c1 * w1[j * 2 + 1] + b1[j], 0.f);
#pragma unroll
    for (int k = 0; k < HEADS; ++k) acc[k] += h * w2[k * 512 + j];
  }
#pragma unroll
  for (int k = 0; k < HEADS; ++k) b169[t * HEADS + k] = acc[k] + b2[k];
}

// ============ prep 3: padded+transposed bias+mask table =====================
// cmb[w][h][j][i]: i>=49 -> 0 ; j>=49 -> -1e30 ; else bias[h,i,j] + mask[w,i,j]
// Transposed (j major, i contiguous) so the score-WMMA C operand (rows
// contiguous per accumulator lane) loads as 2x global_load_b128 per tile.
__global__ __launch_bounds__(256) void combine_bias_mask(
    const float* __restrict__ b169, const int* __restrict__ rel_index,
    const float* __restrict__ mask, float* __restrict__ cmb) {
  int s = blockIdx.x * 256 + threadIdx.x;      // 64*4*64*64 = 1M
  int i = s & 63, j = (s >> 6) & 63, h = (s >> 12) & 3, w = s >> 14;
  float v;
  if (i >= L_TOK) v = 0.f;
  else if (j >= L_TOK) v = -1e30f;
  else v = b169[rel_index[i * L_TOK + j] * HEADS + h] +
           mask[((size_t)w * L_TOK + i) * L_TOK + j];
  cmb[s] = v;
}

// ============ main fused window-attention kernel ============================
__global__ __launch_bounds__(256) void win_attn_main(
    const float* __restrict__ x, const float* __restrict__ scale,
    const float* __restrict__ qkv_b, const float* __restrict__ proj_b,
    const _Float16* __restrict__ wq, const _Float16* __restrict__ wp,
    const float* __restrict__ cmb, float* __restrict__ out) {
  extern __shared__ char smem[];
  const int tid = threadIdx.x;
  const int lane = tid & 31;
  const int wave = tid >> 5;
  const int win = blockIdx.x;
  const int mwin = win & (NW - 1);

  float* qk  = (float*)(smem + OFF_QK);
  float* Sp  = (float*)(smem + OFF_S);
  float* of  = (float*)(smem + OFF_OUTF);
  float* scl = (float*)(smem + OFF_SCALES);    // [2][4][64]

  // ---- phase 0: x[49][128] -> f16 WMMA A-fragments ------------------------
#pragma unroll
  for (int it = 0; it < 2; ++it) {
    int s = tid + it * 256;                    // 16 frags * 32 slots
    int frag = s >> 5, ln = s & 31;
    int mt = frag >> 2, kc = frag & 3;
    int row = mt * 16 + (ln & 15), hi = ln >> 4;
    v16h t;
    if (row < L_TOK) {
      const float* src = x + ((size_t)win * L_TOK + row) * DIM + kc * 32 + hi * 8;
      float4 a0 = *(const float4*)(src);
      float4 a1 = *(const float4*)(src + 4);
      float4 b0 = *(const float4*)(src + 16);
      float4 b1 = *(const float4*)(src + 20);
      t[0]=(_Float16)a0.x; t[1]=(_Float16)a0.y; t[2]=(_Float16)a0.z; t[3]=(_Float16)a0.w;
      t[4]=(_Float16)a1.x; t[5]=(_Float16)a1.y; t[6]=(_Float16)a1.z; t[7]=(_Float16)a1.w;
      t[8]=(_Float16)b0.x; t[9]=(_Float16)b0.y; t[10]=(_Float16)b0.z; t[11]=(_Float16)b0.w;
      t[12]=(_Float16)b1.x; t[13]=(_Float16)b1.y; t[14]=(_Float16)b1.z; t[15]=(_Float16)b1.w;
    } else {
#pragma unroll
      for (int j = 0; j < 16; ++j) t[j] = (_Float16)0.f;
    }
    *(v16h*)(smem + OFF_XA + (frag << 10) + (ln << 5)) = t;
  }
  __syncthreads();

  // ---- phase 1: QKV GEMM 64x384x128; v-tiles stream straight into B-frags -
  for (int nn = 0; nn < 3; ++nn) {
    int n = wave * 3 + nn;                     // 24 N-tiles over 8 waves
    float bcol = qkv_b[n * 16 + (lane & 15)];
#pragma unroll
    for (int m = 0; m < 4; ++m) {
      v8f acc = {};
#pragma unroll
      for (int kc = 0; kc < 4; ++kc) {
        v16h a = *(const v16h*)(smem + OFF_XA + ((m * 4 + kc) << 10) + (lane << 5));
        v16h b = *(const v16h*)(wq + ((size_t)(n * 4 + kc) << 9) + (lane << 4));
        acc = wmma16(a, b, acc);
      }
      if (n < 16) {                            // q,k rows -> f32 LDS (row major)
        int ib = m * 16 + ((lane & 16) ? 8 : 0);
        int j = n * 16 + (lane & 15);
#pragma unroll
        for (int r = 0; r < 8; ++r) qk[(ib + r) * QK_STR + j] = acc[r] + bcol;
      } else {                                 // v: 8 accs == half a B-frag lane slot
        int f = n - 16;                        // feature block 0..7
        int h = f >> 1, nidx = f & 1;
        int frag = h * 4 + (m >> 1) * 2 + nidx;
        _Float16 tmp[8] __attribute__((aligned(16)));
#pragma unroll
        for (int r = 0; r < 8; ++r) tmp[r] = (_Float16)(acc[r] + bcol);
        *(float4*)(smem + OFF_VB + (frag << 10) + (lane << 5) + ((m & 1) << 4)) =
            *(float4*)tmp;
      }
    }
  }
  __syncthreads();

  // ---- phase 2a: inverse norms; exp(-scale[h]) folded into the q scale ----
#pragma unroll
  for (int it = 0; it < 2; ++it) {
    int t = tid + it * 256;                    // (q|k) x 4h x 64tok
    int tok = t & 63, head = (t >> 6) & 3, isK = t >> 8;
    const float4* p = (const float4*)(qk + tok * QK_STR + isK * 128 + head * 32);
    float ss = 0.f;
#pragma unroll
    for (int d = 0; d < 8; ++d) {
      float4 v = p[d];
      ss += v.x * v.x + v.y * v.y + v.z * v.z + v.w * v.w;
    }
    float r = 1.f / fmaxf(sqrtf(ss), 1e-12f);
    if (!isK) r *= __expf(-scale[head]);
    scl[isK * 256 + head * 64 + tok] = r;
  }
  __syncthreads();

  // ---- phase 2b: q-hat A-frags (pad rows zeroed) + k-hat B-frags ----------
#pragma unroll
  for (int it = 0; it < 2; ++it) {             // q-hat
    int s = tid + it * 256;
    int frag = s >> 5, ln = s & 31;
    int h = frag >> 2, m = frag & 3;
    int tok = m * 16 + (ln & 15), hi = ln >> 4;
    v16h t;
    if (tok < L_TOK) {
      float sq = scl[h * 64 + tok];
      const float4* p = (const float4*)(qk + tok * QK_STR + h * 32 + hi * 8);
      float4 a0 = p[0], a1 = p[1], b0 = p[4], b1 = p[5];
      t[0]=(_Float16)(a0.x*sq); t[1]=(_Float16)(a0.y*sq); t[2]=(_Float16)(a0.z*sq); t[3]=(_Float16)(a0.w*sq);
      t[4]=(_Float16)(a1.x*sq); t[5]=(_Float16)(a1.y*sq); t[6]=(_Float16)(a1.z*sq); t[7]=(_Float16)(a1.w*sq);
      t[8]=(_Float16)(b0.x*sq); t[9]=(_Float16)(b0.y*sq); t[10]=(_Float16)(b0.z*sq); t[11]=(_Float16)(b0.w*sq);
      t[12]=(_Float16)(b1.x*sq); t[13]=(_Float16)(b1.y*sq); t[14]=(_Float16)(b1.z*sq); t[15]=(_Float16)(b1.w*sq);
    } else {
#pragma unroll
      for (int j = 0; j < 16; ++j) t[j] = (_Float16)0.f;
    }
    *(v16h*)(smem + OFF_QA + (frag << 10) + (ln << 5)) = t;
  }
#pragma unroll
  for (int it = 0; it < 2; ++it) {             // k-hat
    int s = tid + it * 256;
    int frag = s >> 5, ln = s & 31;
    int h = frag >> 2, n = frag & 3;
    int tok = n * 16 + (ln & 15), hi = ln >> 4;
    float sk = scl[256 + h * 64 + tok];
    const float4* p = (const float4*)(qk + tok * QK_STR + 128 + h * 32 + hi * 8);
    float4 a0 = p[0], a1 = p[1], b0 = p[4], b1 = p[5];
    v16h t;
    t[0]=(_Float16)(a0.x*sk); t[1]=(_Float16)(a0.y*sk); t[2]=(_Float16)(a0.z*sk); t[3]=(_Float16)(a0.w*sk);
    t[4]=(_Float16)(a1.x*sk); t[5]=(_Float16)(a1.y*sk); t[6]=(_Float16)(a1.z*sk); t[7]=(_Float16)(a1.w*sk);
    t[8]=(_Float16)(b0.x*sk); t[9]=(_Float16)(b0.y*sk); t[10]=(_Float16)(b0.z*sk); t[11]=(_Float16)(b0.w*sk);
    t[12]=(_Float16)(b1.x*sk); t[13]=(_Float16)(b1.y*sk); t[14]=(_Float16)(b1.z*sk); t[15]=(_Float16)(b1.w*sk);
    *(v16h*)(smem + OFF_KB + (frag << 10) + (ln << 5)) = t;
  }
  __syncthreads();

  // ---- phase 3: S = q-hat @ k-hat^T + C(bias+mask+pad) via WMMA C operand -
  {
    int h = wave >> 1;
    const float* cb = cmb + (((size_t)mwin * HEADS + h) << 12);  // [64j][64i]
#pragma unroll
    for (int idx = 0; idx < 8; ++idx) {
      int tile = (wave & 1) * 8 + idx;
      int m = tile >> 2, n = tile & 3;
      v16h a = *(const v16h*)(smem + OFF_QA + ((h * 4 + m) << 10) + (lane << 5));
      v16h b = *(const v16h*)(smem + OFF_KB + ((h * 4 + n) << 10) + (lane << 5));
      int ib = m * 16 + ((lane & 16) ? 8 : 0);
      int jc = n * 16 + (lane & 15);
      const float* cp = cb + (size_t)jc * 64 + ib;
      float4 c0 = *(const float4*)(cp);
      float4 c1 = *(const float4*)(cp + 4);
      v8f c;
      c[0]=c0.x; c[1]=c0.y; c[2]=c0.z; c[3]=c0.w;
      c[4]=c1.x; c[5]=c1.y; c[6]=c1.z; c[7]=c1.w;
      v8f acc = wmma16(a, b, c);
#pragma unroll
      for (int r = 0; r < 8; ++r) Sp[(h * 64 + ib + r) * 64 + jc] = acc[r];
    }
  }
  __syncthreads();

  // ---- phase 4: row softmax over 49 cols (float4-vectorized) --------------
  if (tid < HEADS * L_TOK) {
    int h = tid / L_TOK, i = tid % L_TOK;
    float* rp = Sp + (h * 64 + i) * 64;
    float4* rp4 = (float4*)rp;
    float mx = rp[48];
    for (int j4 = 0; j4 < 12; ++j4) {
      float4 v = rp4[j4];
      mx = fmaxf(mx, fmaxf(fmaxf(v.x, v.y), fmaxf(v.z, v.w)));
    }
    float sum = 0.f;
    for (int j4 = 0; j4 < 12; ++j4) {
      float4 v = rp4[j4];
      v.x = __expf(v.x - mx); v.y = __expf(v.y - mx);
      v.z = __expf(v.z - mx); v.w = __expf(v.w - mx);
      sum += v.x + v.y + v.z + v.w;
      rp4[j4] = v;
    }
    float e48 = __expf(rp[48] - mx);
    sum += e48;
    float inv = 1.f / sum;
    for (int j4 = 0; j4 < 12; ++j4) {
      float4 v = rp4[j4];
      v.x *= inv; v.y *= inv; v.z *= inv; v.w *= inv;
      rp4[j4] = v;
    }
    float4 tail = {e48 * inv, 0.f, 0.f, 0.f};
    rp4[12] = tail;
    float4 z = {0.f, 0.f, 0.f, 0.f};
    rp4[13] = z; rp4[14] = z; rp4[15] = z;
  }
  __syncthreads();

  // ---- phase 5: attn -> f16 A-fragments [h][m][kc] ------------------------
#pragma unroll
  for (int it = 0; it < 4; ++it) {
    int s = tid + it * 256;                    // 32 frags * 32 slots
    int frag = s >> 5, ln = s & 31;
    int kc = frag & 1, m = (frag >> 1) & 3, h = frag >> 3;
    int ti = m * 16 + (ln & 15), hi = ln >> 4;
    const float4* p = (const float4*)(Sp + (h * 64 + ti) * 64 + kc * 32 + hi * 8);
    float4 a0 = p[0], a1 = p[1], b0 = p[4], b1 = p[5];
    v16h t;
    t[0]=(_Float16)a0.x; t[1]=(_Float16)a0.y; t[2]=(_Float16)a0.z; t[3]=(_Float16)a0.w;
    t[4]=(_Float16)a1.x; t[5]=(_Float16)a1.y; t[6]=(_Float16)a1.z; t[7]=(_Float16)a1.w;
    t[8]=(_Float16)b0.x; t[9]=(_Float16)b0.y; t[10]=(_Float16)b0.z; t[11]=(_Float16)b0.w;
    t[12]=(_Float16)b1.x; t[13]=(_Float16)b1.y; t[14]=(_Float16)b1.z; t[15]=(_Float16)b1.w;
    *(v16h*)(smem + OFF_AATTN + (frag << 10) + (ln << 5)) = t;
  }
  __syncthreads();

  // ---- phase 6: P = attn @ v ----------------------------------------------
  {
    int h = wave >> 1, n = wave & 1;
#pragma unroll
    for (int m = 0; m < 4; ++m) {
      v8f acc = {};
#pragma unroll
      for (int kc = 0; kc < 2; ++kc) {
        v16h a = *(const v16h*)(smem + OFF_AATTN + (((h * 4 + m) * 2 + kc) << 10) + (lane << 5));
        v16h b = *(const v16h*)(smem + OFF_VB + ((h * 4 + kc * 2 + n) << 10) + (lane << 5));
        acc = wmma16(a, b, acc);
      }
      int ib = m * 16 + ((lane & 16) ? 8 : 0);
      int col = h * 32 + n * 16 + (lane & 15);
#pragma unroll
      for (int r = 0; r < 8; ++r) of[(ib + r) * OUTF_STR + col] = acc[r];
    }
  }
  __syncthreads();

  // ---- phase 7: P -> f16 A-fragments for proj (reuse QA region) -----------
#pragma unroll
  for (int it = 0; it < 2; ++it) {
    int s = tid + it * 256;                    // 16 frags * 32 slots
    int frag = s >> 5, ln = s & 31;
    int m = frag >> 2, kc = frag & 3;
    int row = m * 16 + (ln & 15), hi = ln >> 4;
    const float4* p = (const float4*)(of + row * OUTF_STR + kc * 32 + hi * 8);
    float4 a0 = p[0], a1 = p[1], b0 = p[4], b1 = p[5];
    v16h t;
    t[0]=(_Float16)a0.x; t[1]=(_Float16)a0.y; t[2]=(_Float16)a0.z; t[3]=(_Float16)a0.w;
    t[4]=(_Float16)a1.x; t[5]=(_Float16)a1.y; t[6]=(_Float16)a1.z; t[7]=(_Float16)a1.w;
    t[8]=(_Float16)b0.x; t[9]=(_Float16)b0.y; t[10]=(_Float16)b0.z; t[11]=(_Float16)b0.w;
    t[12]=(_Float16)b1.x; t[13]=(_Float16)b1.y; t[14]=(_Float16)b1.z; t[15]=(_Float16)b1.w;
    *(v16h*)(smem + OFF_QA + (frag << 10) + (ln << 5)) = t;
  }
  __syncthreads();

  // ---- phase 8: out = P @ proj_w^T + proj_b -------------------------------
  {
    int n = wave;
    float pb = proj_b[n * 16 + (lane & 15)];
#pragma unroll
    for (int m = 0; m < 4; ++m) {
      v8f acc = {};
#pragma unroll
      for (int kc = 0; kc < 4; ++kc) {
        v16h a = *(const v16h*)(smem + OFF_QA + ((m * 4 + kc) << 10) + (lane << 5));
        v16h b = *(const v16h*)(wp + ((size_t)(n * 4 + kc) << 9) + (lane << 4));
        acc = wmma16(a, b, acc);
      }
      int ib = m * 16 + ((lane & 16) ? 8 : 0);
      int j = n * 16 + (lane & 15);
#pragma unroll
      for (int r = 0; r < 8; ++r) {
        int i = ib + r;
        if (i < L_TOK) out[((size_t)win * L_TOK + i) * DIM + j] = acc[r] + pb;
      }
    }
  }
}

extern "C" void kernel_launch(void* const* d_in, const int* in_sizes, int n_in,
                              void* d_out, int out_size, void* d_ws, size_t ws_size,
                              hipStream_t stream) {
  const float* x          = (const float*)d_in[0];
  const float* mask       = (const float*)d_in[1];
  const float* rel_coords = (const float*)d_in[2];
  const float* scale      = (const float*)d_in[3];
  const float* rpb_w1     = (const float*)d_in[4];
  const float* rpb_b1     = (const float*)d_in[5];
  const float* rpb_w2     = (const float*)d_in[6];
  const float* rpb_b2     = (const float*)d_in[7];
  const float* qkv_w      = (const float*)d_in[8];
  const float* qkv_b      = (const float*)d_in[9];
  const float* proj_w     = (const float*)d_in[10];
  const float* proj_b     = (const float*)d_in[11];
  const int*   rel_index  = (const int*)d_in[12];

  char* ws = (char*)d_ws;
  _Float16* wq   = (_Float16*)(ws + WS_WQKV);
  _Float16* wp   = (_Float16*)(ws + WS_WPROJ);
  float*    b169 = (float*)(ws + WS_B169);
  float*    cmb  = (float*)(ws + WS_CMB);

  pack_weights<<<16, 256, 0, stream>>>(qkv_w, proj_w, wq, wp);
  rpb_mlp<<<1, 256, 0, stream>>>(rel_coords, rpb_w1, rpb_b1, rpb_w2, rpb_b2, b169);
  combine_bias_mask<<<4096, 256, 0, stream>>>(b169, rel_index, mask, cmb);
  win_attn_main<<<8192, 256, SMEM_BYTES, stream>>>(x, scale, qkv_b, proj_b,
                                                   wq, wp, cmb, (float*)d_out);
}
